// AttnDecoderRNN_87093346828826
// MI455X (gfx1250) — matrix operand order repaired and checked by reading, not compiled, not run on hardware
//
#include <hip/hip_runtime.h>
#include <hip/hip_bf16.h>

// ---------------------------------------------------------------------------
// AttnDecoderRNN for MI455X (gfx1250, wave32, WMMA + async global->LDS).
// Core ops: v_wmma_f32_16x16x32_f16, global_load_async_to_lds_b128.
// ---------------------------------------------------------------------------

typedef __attribute__((ext_vector_type(16))) _Float16 v16h;
typedef __attribute__((ext_vector_type(8)))  float    v8f;
typedef __attribute__((ext_vector_type(4)))  _Float16 v4h;

union Frag16 { v16h h; unsigned u[8]; };

#define TM 128
#define TN 64
#define TK 32
#define APAD 4      // a_sh row pad (halves)
#define WPAD 8      // w_sh row pad (halves): 80B row stride, keeps 16B alignment

// C[z] = act( [A1|A2] * [W1|W2]^T + bias[z] + bias2 )
// A1:[M,K1] A2:[M,K2] fp32; W1:[N,K1] W2:[N,K2] fp16 (pre-converted).
// N = gridDim.x*64, K1,K2 multiples of 32 (K2 may be 0 -> A2/W2 untouched).
__global__ __launch_bounds__(256)
void gemm_wmma_kernel(const float* __restrict__ A1, long lda1, long strideAz,
                      const float* __restrict__ A2, long lda2,
                      const _Float16* __restrict__ W1, int ldw1, long strideWz,
                      const _Float16* __restrict__ W2, int ldw2,
                      const float* __restrict__ bias, long strideBz,
                      const float* __restrict__ bias2,
                      float* __restrict__ C, long ldc, long strideCz,
                      int M, int K1, int K2, int act)
{
    __shared__ _Float16 a_sh[TM][TK + APAD];
    __shared__ _Float16 w_sh[2][TN][TK + WPAD];   // double-buffered, async-filled

    const int tid  = threadIdx.x;
    const int lane = tid & 31;             // wave32
    const int wave = tid >> 5;             // 8 waves * 16 rows = 128 M rows
    const int m0   = blockIdx.y * TM;
    const int n0   = blockIdx.x * TN;
    const int z    = blockIdx.z;

    A1 += (long)z * strideAz;
    W1 += (long)z * strideWz;
    C  += (long)z * strideCz;
    if (bias) bias += (long)z * strideBz;

    const int K = K1 + K2;

    // A staging coords: 8 float4-quads per 32-wide row
    const int arow0 = tid >> 3;            // rows arow0 + {0,32,64,96}
    const int kq    = (tid & 7) << 2;
    // W async coords: one b128 (8 halves) per thread -> 64 x 32 halves per tile
    const int wrow  = tid >> 2;            // 0..63
    const int wseg  = (tid & 3) << 3;      // half-offset 0,8,16,24 (16B aligned)

    v8f acc[4] = {};
    float4 ra[4];

    auto load_A_regs = [&](int k0) {
        const float* Ap; long lda; int kk;
        if (k0 < K1) { Ap = A1; lda = lda1; kk = k0; }
        else         { Ap = A2; lda = lda2; kk = k0 - K1; }
        #pragma unroll
        for (int i = 0; i < 4; ++i) {
            int m = m0 + arow0 + i * 32;
            if (m > M - 1) m = M - 1;      // clamp: no divergence, store is guarded
            ra[i] = *(const float4*)(Ap + (long)m * lda + kk + kq);
        }
    };
    // async copy of one 64x32 fp16 weight tile into w_sh[ib] (ASYNCcnt path)
    auto issue_async_w = [&](int k0, int ib) {
        const _Float16* Wp; int ldw; int kk;
        if (k0 < K1) { Wp = W1; ldw = ldw1; kk = k0; }
        else         { Wp = W2; ldw = ldw2; kk = k0 - K1; }
        const _Float16* gp = Wp + (long)(n0 + wrow) * ldw + kk + wseg;
        unsigned lds = (unsigned)(unsigned long)&w_sh[ib][wrow][wseg];
        asm volatile("global_load_async_to_lds_b128 %0, %1, off"
                     :: "v"(lds), "v"(gp) : "memory");
    };

    const int half_sel = lane >> 4;
    const int sub      = lane & 15;

    load_A_regs(0);
    issue_async_w(0, 0);

    int ib = 0;
    for (int k0 = 0; k0 < K; k0 += TK, ib ^= 1) {
        // ---- drain staged A registers into LDS (fp32 -> fp16) ----
        #pragma unroll
        for (int i = 0; i < 4; ++i) {
            _Float16* d = &a_sh[arow0 + i * 32][kq];
            d[0] = (_Float16)ra[i].x; d[1] = (_Float16)ra[i].y;
            d[2] = (_Float16)ra[i].z; d[3] = (_Float16)ra[i].w;
        }
        // wait for this wave's async weight-tile fill of w_sh[ib]
        asm volatile("s_wait_asynccnt 0x0" ::: "memory");
        __syncthreads();

        // ---- overlap next tile's traffic with this tile's WMMAs ----
        if (k0 + TK < K) {
            issue_async_w(k0 + TK, ib ^ 1);
            load_A_regs(k0 + TK);
        }

        // ---- A fragment: 16x32, lane-half holds k in two 8-blocks (ISA 7.12.2) ----
        Frag16 af;
        const int mrow = (wave << 4) + sub;
        #pragma unroll
        for (int p = 0; p < 8; ++p) {
            int kk = ((p & 4) << 2) + (half_sel << 3) + ((p & 3) << 1);
            af.u[p] = *(const unsigned*)&a_sh[mrow][kk];
        }
        // ---- all 4 B fragments, then chained WMMAs ----
        Frag16 bf[4];
        #pragma unroll
        for (int j = 0; j < 4; ++j) {
            const int nrow = (j << 4) + sub;
            #pragma unroll
            for (int p = 0; p < 8; ++p) {
                int kk = ((p & 4) << 2) + (half_sel << 3) + ((p & 3) << 1);
                bf[j].u[p] = *(const unsigned*)&w_sh[ib][nrow][kk];
            }
        }
        #pragma unroll
        for (int j = 0; j < 4; ++j)
            acc[j] = __builtin_amdgcn_wmma_f32_16x16x32_f16(
                         false, af.h, false, bf[j].h, (short)0, acc[j], false, false);
        __syncthreads();
    }

    // ---- epilogue: C/D layout = VGPR r -> M = r + (lane>=16)*8, N = lane&15 ----
    const int rbase = half_sel << 3;
    #pragma unroll
    for (int j = 0; j < 4; ++j) {
        int n = n0 + (j << 4) + sub;
        float bv = 0.f;
        if (bias)  bv += bias[n];
        if (bias2) bv += bias2[n];
        #pragma unroll
        for (int r = 0; r < 8; ++r) {
            int m = m0 + (wave << 4) + rbase + r;
            if (m < M) {
                float v = acc[j][r] + bv;
                if (act) v = fmaxf(v, 0.f);
                C[(long)m * ldc + n] = v;
            }
        }
    }
}

// ---------------------------------------------------------------------------
// one-time fp32 -> fp16 weight conversion (float4 in, 4 halves out)
__global__ void cvt_f32_to_f16_kernel(const float* __restrict__ src,
                                      _Float16* __restrict__ dst, long nq)
{
    long i = (long)blockIdx.x * blockDim.x + threadIdx.x;
    if (i >= nq) return;
    float4 v = ((const float4*)src)[i];
    v4h o;
    o.x = (_Float16)v.x; o.y = (_Float16)v.y;
    o.z = (_Float16)v.z; o.w = (_Float16)v.w;
    ((v4h*)dst)[i] = o;
}

__device__ __forceinline__ float sigmoidf_(float x) { return 1.f / (1.f + __expf(-x)); }

// g:[B,4H] gate order i,f,g,o  ->  update c,h ; also scatter h into hout[:,t,:]
__global__ void lstm_cell_kernel(const float* __restrict__ g,
                                 float* __restrict__ c, float* __restrict__ h,
                                 float* __restrict__ hout,
                                 int t, int Bn, int Hh, int Tn)
{
    int idx = blockIdx.x * blockDim.x + threadIdx.x;
    if (idx >= Bn * Hh) return;
    int b = idx / Hh, hh = idx - b * Hh;
    const float* gr = g + (long)b * 4 * Hh;
    float iv = sigmoidf_(gr[hh]);
    float fv = sigmoidf_(gr[Hh + hh]);
    float gv = tanhf(gr[2 * Hh + hh]);
    float ov = sigmoidf_(gr[3 * Hh + hh]);
    float cv = fv * c[idx] + iv * gv;
    float hv = ov * tanhf(cv);
    c[idx] = cv;
    h[idx] = hv;
    hout[((long)b * Tn + t) * Hh + hh] = hv;
}

// softmax over 256 features per (b,t) row, then applied = w * x. One wave/row.
__global__ __launch_bounds__(256)
void softmax_apply_kernel(const float* __restrict__ attn,
                          const float* __restrict__ x,
                          float* __restrict__ out, int rows)
{
    int wave = threadIdx.x >> 5, lane = threadIdx.x & 31;
    int row  = blockIdx.x * 8 + wave;
    if (row >= rows) return;
    const float* a = attn + (long)row * 256;
    float v[8], mx = -3.0e38f;
    #pragma unroll
    for (int i = 0; i < 8; ++i) { v[i] = a[lane + i * 32]; mx = fmaxf(mx, v[i]); }
    #pragma unroll
    for (int off = 16; off; off >>= 1) mx = fmaxf(mx, __shfl_xor(mx, off, 32));
    float s = 0.f;
    #pragma unroll
    for (int i = 0; i < 8; ++i) { v[i] = __expf(v[i] - mx); s += v[i]; }
    #pragma unroll
    for (int off = 16; off; off >>= 1) s += __shfl_xor(s, off, 32);
    float inv = 1.f / s;
    const float* xr = x + (long)row * 256;
    float* o = out + (long)row * 256;
    #pragma unroll
    for (int i = 0; i < 8; ++i) o[lane + i * 32] = v[i] * inv * xr[lane + i * 32];
}

// x[b,t,:] = emb[id[b,t],:]  (float4 vectorized, E=256 -> 64 quads/row)
__global__ void embed_kernel(const int* __restrict__ ids,
                             const float* __restrict__ emb,
                             float* __restrict__ x, long totalq, int Eq)
{
    long i = (long)blockIdx.x * blockDim.x + threadIdx.x;
    if (i >= totalq) return;
    long row = i / Eq;
    int  eq  = (int)(i - row * Eq);
    ((float4*)x)[i] = ((const float4*)emb)[(long)ids[row] * Eq + eq];
}

// tag[b] = d3[b,:] . Wout + bout
__global__ void final_dot_kernel(const float* __restrict__ d3,
                                 const float* __restrict__ Wout,
                                 const float* __restrict__ bout,
                                 float* __restrict__ tag, int Bn)
{
    int b = blockIdx.x * blockDim.x + threadIdx.x;
    if (b >= Bn) return;
    const float* r = d3 + (long)b * 64;
    float s = bout[0];
    #pragma unroll 8
    for (int i = 0; i < 64; ++i) s += r[i] * Wout[i];
    tag[b] = s;
}

// ---------------------------------------------------------------------------
extern "C" void kernel_launch(void* const* d_in, const int* in_sizes, int n_in,
                              void* d_out, int out_size, void* d_ws, size_t ws_size,
                              hipStream_t stream)
{
    (void)in_sizes; (void)n_in; (void)out_size; (void)ws_size;
    const int B = 1000, T = 101, H = 256, E = 256, FH = 1024;

    const int*   input_ids = (const int*)  d_in[0];
    const float* emb   = (const float*)d_in[1];
    const float* Wih1  = (const float*)d_in[2];
    const float* Whh1  = (const float*)d_in[3];
    const float* bih1  = (const float*)d_in[4];
    const float* bhh1  = (const float*)d_in[5];
    const float* Wih2  = (const float*)d_in[6];
    const float* Whh2  = (const float*)d_in[7];
    const float* bih2  = (const float*)d_in[8];
    const float* bhh2  = (const float*)d_in[9];
    const float* Wattn = (const float*)d_in[10];
    const float* battn = (const float*)d_in[11];
    const float* W1    = (const float*)d_in[12];
    const float* b1    = (const float*)d_in[13];
    const float* W2    = (const float*)d_in[14];
    const float* b2    = (const float*)d_in[15];
    const float* W3    = (const float*)d_in[16];
    const float* b3    = (const float*)d_in[17];
    const float* Wout  = (const float*)d_in[18];
    const float* bout  = (const float*)d_in[19];

    // Workspace layout (fp32 region then fp16 weight region); total ~361 MB.
    float* ws = (float*)d_ws;
    const long BTE = (long)B * T * E;                 // 25,856,000
    long off = 0;
    float* x     = ws + off; off += BTE;
    float* h1    = ws + off; off += BTE;              // reused as attn after LSTM2
    float* h2    = ws + off; off += BTE;
    float* g     = ws + off; off += (long)B * FH;
    float* h_cur = ws + off; off += (long)B * H;
    float* c_cur = ws + off; off += (long)B * H;
    float* d1    = ws + off; off += (long)B * 512;
    float* d2    = ws + off; off += (long)B * 128;
    float* d3    = ws + off; off += (long)B * 64;
    float* attn  = h1;

    _Float16* hbase = (_Float16*)(ws + off);          // 16B-aligned (off is even)
    long ho = 0;
    _Float16* Wih1h = hbase + ho; ho += (long)FH * H;
    _Float16* Whh1h = hbase + ho; ho += (long)FH * H;
    _Float16* Wih2h = hbase + ho; ho += (long)FH * H;
    _Float16* Whh2h = hbase + ho; ho += (long)FH * H;
    _Float16* Wattnh= hbase + ho; ho += (long)T * H * H;
    _Float16* W1h   = hbase + ho; ho += (long)512 * T * E;
    _Float16* W2h   = hbase + ho; ho += (long)128 * 512;
    _Float16* W3h   = hbase + ho; ho += (long)64 * 128;

    float* tag     = (float*)d_out;                   // [B]
    float* applied = (float*)d_out + B;               // [B,T,E]

    auto cvt = [&](const float* s, _Float16* d, long n) {
        long nq = n / 4;
        cvt_f32_to_f16_kernel<<<(unsigned)((nq + 255) / 256), 256, 0, stream>>>(s, d, nq);
    };
    auto gemm = [&](const float* A1, long lda1, long sAz,
                    const float* A2, long lda2,
                    const _Float16* Wt1, int ldw1, long sWz,
                    const _Float16* Wt2, int ldw2,
                    const float* bias, long sBz, const float* bias2,
                    float* Cp, long ldc, long sCz,
                    int M, int N, int K1, int K2, int act, int Z) {
        dim3 grid(N / TN, (M + TM - 1) / TM, Z);
        gemm_wmma_kernel<<<grid, 256, 0, stream>>>(A1, lda1, sAz, A2, lda2,
                                                   Wt1, ldw1, sWz, Wt2, ldw2,
                                                   bias, sBz, bias2,
                                                   Cp, ldc, sCz, M, K1, K2, act);
    };

    // 0) One-time fp32 -> fp16 weight conversion (async-copy friendly)
    cvt(Wih1,  Wih1h,  (long)FH * H);
    cvt(Whh1,  Whh1h,  (long)FH * H);
    cvt(Wih2,  Wih2h,  (long)FH * H);
    cvt(Whh2,  Whh2h,  (long)FH * H);
    cvt(Wattn, Wattnh, (long)T * H * H);
    cvt(W1,    W1h,    (long)512 * T * E);
    cvt(W2,    W2h,    (long)128 * 512);
    cvt(W3,    W3h,    (long)64 * 128);

    // 1) Embedding gather (float4)
    {
        long totalq = BTE / 4;
        embed_kernel<<<(unsigned)((totalq + 255) / 256), 256, 0, stream>>>(
            input_ids, emb, x, totalq, E / 4);
    }

    // 2) Two stacked LSTM layers, 101 sequential steps each.
    //    One split-K GEMM per step: g = [x_t | h] @ [Wih | Whh]^T + bih + bhh
    const _Float16* WihL[2] = {Wih1h, Wih2h};
    const _Float16* WhhL[2] = {Whh1h, Whh2h};
    const float*    bihL[2] = {bih1, bih2};
    const float*    bhhL[2] = {bhh1, bhh2};
    for (int layer = 0; layer < 2; ++layer) {
        const float* in  = layer ? h1 : x;
        float*       out = layer ? h2 : h1;
        hipMemsetAsync(h_cur, 0, (size_t)B * H * sizeof(float), stream);
        hipMemsetAsync(c_cur, 0, (size_t)B * H * sizeof(float), stream);
        for (int t = 0; t < T; ++t) {
            gemm(in + (long)t * H, (long)T * H, 0,      // A1 = x_t (strided rows)
                 h_cur, H,                              // A2 = h_{t-1}
                 WihL[layer], H, 0,                     // W1 (fp16)
                 WhhL[layer], H,                        // W2 (fp16)
                 bihL[layer], 0, bhhL[layer],
                 g, FH, 0, B, FH, H, H, 0, 1);
            lstm_cell_kernel<<<(B * H + 255) / 256, 256, 0, stream>>>(
                g, c_cur, h_cur, out, t, B, H, T);
        }
    }

    // 3) Attention: per-t weights, one z-batched WMMA GEMM (z = t)
    gemm(h2, (long)T * H, H, nullptr, 0,
         Wattnh, H, (long)H * H, nullptr, 0,
         battn, H, nullptr,
         attn, (long)T * H, H,
         B, H, H, 0, 0, T);

    // 4) softmax over features, applied = w * x  (written straight into d_out)
    softmax_apply_kernel<<<(B * T + 7) / 8, 256, 0, stream>>>(attn, x, applied, B * T);

    // 5) MLP head
    gemm(applied, (long)T * E, 0, nullptr, 0,
         W1h, T * E, 0, nullptr, 0, b1, 0, nullptr,
         d1, 512, 0, B, 512, T * E, 0, 1, 1);
    gemm(d1, 512, 0, nullptr, 0,
         W2h, 512, 0, nullptr, 0, b2, 0, nullptr,
         d2, 128, 0, B, 128, 512, 0, 1, 1);
    gemm(d2, 128, 0, nullptr, 0,
         W3h, 128, 0, nullptr, 0, b3, 0, nullptr,
         d3, 64, 0, B, 64, 128, 0, 1, 1);
    final_dot_kernel<<<(B + 255) / 256, 256, 0, stream>>>(d3, Wout, bout, tag, B);
}
